// GATBlock_87342454931670
// MI455X (gfx1250) — matrix operand order repaired and checked by reading, not compiled, not run on hardware
//
#include <hip/hip_runtime.h>
#include <hip/hip_bf16.h>

#define N_NODES 50000
#define N_EDGES 800000
#define IN_DIM  512
#define HID     256
#define LAT     64
#define SLOPE   0.2f

typedef __attribute__((ext_vector_type(16))) __bf16 v16bf;
typedef __attribute__((ext_vector_type(8)))  __bf16 v8bf;
typedef __attribute__((ext_vector_type(8)))  float  v8f;
typedef __attribute__((ext_vector_type(4)))  float  v4f;
typedef __attribute__((ext_vector_type(4)))  int    v4i;

#define LDS_STRIDE 40  // 80 bytes/row: 16B aligned, conflict-free across 16 rows

#if defined(__gfx1250__) && __has_builtin(__builtin_amdgcn_global_load_async_to_lds_b128)
#define HAVE_ASYNC_LDS 1
__device__ __forceinline__ void async_b128(const void* g, void* l) {
  __builtin_amdgcn_global_load_async_to_lds_b128((v4i*)g, (v4i*)l, 0, 0);
}
template <int N> __device__ __forceinline__ void wait_async() {
#if __has_builtin(__builtin_amdgcn_s_wait_asynccnt)
  __builtin_amdgcn_s_wait_asynccnt(N);
#else
  if (N == 0) asm volatile("s_wait_asynccnt 0x0" ::: "memory");
  else        asm volatile("s_wait_asynccnt 0x5" ::: "memory");
#endif
}
#else
#define HAVE_ASYNC_LDS 0
#endif

__device__ __forceinline__ v8bf cvt8(v4f a, v4f b) {
  v8bf r;
  r[0] = (__bf16)a[0]; r[1] = (__bf16)a[1]; r[2] = (__bf16)a[2]; r[3] = (__bf16)a[3];
  r[4] = (__bf16)b[0]; r[5] = (__bf16)b[1]; r[6] = (__bf16)b[2]; r[7] = (__bf16)b[3];
  return r;
}

// ---------------------------------------------------------------------------
// bf16 WMMA GEMM: C[M,N](f32) = A[M,K](bf16) * B^T   with B stored [N,K] bf16.
// Block: 256 threads = 8 waves. Block tile 256x64, BK=32.
// Wave w -> rows [32w, 32w+32): 2 A frags x 4 B frags = 8 WMMAs / K-step.
// Staging: double-buffered GLOBAL_LOAD_ASYNC_TO_LDS_B128, pipelined one tile
// ahead (async loads complete in order -> s_wait_asynccnt(5) drains tile k
// while tile k+1 remains in flight over the WMMAs).
// ---------------------------------------------------------------------------
__global__ __launch_bounds__(256)
void gemm_bf16_wmma(const __bf16* __restrict__ A, const __bf16* __restrict__ B,
                    float* __restrict__ C, int M, int N, int K) {
#if HAVE_ASYNC_LDS
  __shared__ __align__(16) __bf16 As[2][256 * LDS_STRIDE];
  __shared__ __align__(16) __bf16 Bs[2][64 * LDS_STRIDE];
#else
  __shared__ __align__(16) __bf16 As[1][256 * LDS_STRIDE];
  __shared__ __align__(16) __bf16 Bs[1][64 * LDS_STRIDE];
#endif

  const int t     = threadIdx.x;
  const int wave  = t >> 5;
  const int lane  = t & 31;
  const int m16   = lane & 15;
  const int khalf = lane >> 4;
  const int blockRow = blockIdx.x * 256;
  const int blockCol = blockIdx.y * 64;

  // A: thread t stages row t of the tile (32 halfs). Clamp pad rows (never stored).
  int gra = blockRow + t;
  gra = gra < M ? gra : (M - 1);
  const __bf16* arow = A + (size_t)gra * K;
  // B: thread t stages 8 halfs of row n = t>>2 at kq = (t&3)*8.
  const int bn = t >> 2, bkq = (t & 3) * 8;
  const __bf16* brow = B + (size_t)(blockCol + bn) * K + bkq;

  v8f acc[2][4] = {};

  auto compute = [&](int buf) {
    union frag { v16bf v; v8bf h[2]; };
    frag af[2];
#pragma unroll
    for (int g = 0; g < 2; ++g) {
      const __bf16* ap = &As[buf][(wave * 32 + g * 16 + m16) * LDS_STRIDE];
      af[g].h[0] = *(const v8bf*)(ap + khalf * 8);        // K = khalf*8 + 0..7
      af[g].h[1] = *(const v8bf*)(ap + 16 + khalf * 8);   // K = 16 + khalf*8 + 0..7
    }
#pragma unroll
    for (int j = 0; j < 4; ++j) {
      frag bfr;
      const __bf16* bp = &Bs[buf][(j * 16 + m16) * LDS_STRIDE + khalf * 8];
      bfr.h[0] = *(const v8bf*)(bp);                      // K = khalf*8 + 0..7
      bfr.h[1] = *(const v8bf*)(bp + 16);                 // K = 16 + khalf*8 + 0..7
      acc[0][j] = __builtin_amdgcn_wmma_f32_16x16x32_bf16(
          false, af[0].v, false, bfr.v, (short)0, acc[0][j], false, false);
      acc[1][j] = __builtin_amdgcn_wmma_f32_16x16x32_bf16(
          false, af[1].v, false, bfr.v, (short)0, acc[1][j], false, false);
    }
  };

#if HAVE_ASYNC_LDS
  auto stage = [&](int k0, int buf) {
#pragma unroll
    for (int q = 0; q < 4; ++q)
      async_b128(arow + k0 + q * 8, &As[buf][t * LDS_STRIDE + q * 8]);
    async_b128(brow + k0, &Bs[buf][bn * LDS_STRIDE + bkq]);
  };

  const int nk = K >> 5;      // K/32 steps (>= 2 for all our GEMMs)
  stage(0, 0);
  for (int ks = 0; ks < nk; ++ks) {
    const int buf = ks & 1;
    if (ks + 2 < nk) __builtin_prefetch(arow + (ks + 2) * 32, 0, 3);
    if (ks + 1 < nk) {
      stage((ks + 1) * 32, buf ^ 1);  // overlap tile k+1 DMA with tile k WMMAs
      wait_async<5>();                // in-order completion: drains tile k only
    } else {
      wait_async<0>();
    }
    __syncthreads();                  // all waves' tile-k data resident in LDS
    compute(buf);
    __syncthreads();                  // buf is re-staged next iteration
  }
#else
  for (int k0 = 0; k0 < K; k0 += 32) {
    if (k0 + 32 < K) __builtin_prefetch(arow + k0 + 32, 0, 3);
#pragma unroll
    for (int q = 0; q < 4; ++q)
      *(v8bf*)(&As[0][t * LDS_STRIDE + q * 8]) = *(const v8bf*)(arow + k0 + q * 8);
    *(v8bf*)(&Bs[0][bn * LDS_STRIDE + bkq]) = *(const v8bf*)(brow + k0);
    __syncthreads();
    compute(0);
    __syncthreads();
  }
#endif

  // Store: lane -> N = blockCol + j*16 + m16 ; VGPR i -> M = khalf*8 + i
#pragma unroll
  for (int g = 0; g < 2; ++g) {
#pragma unroll
    for (int j = 0; j < 4; ++j) {
      int gcol = blockCol + j * 16 + m16;
#pragma unroll
      for (int i = 0; i < 8; ++i) {
        int grow = blockRow + wave * 32 + g * 16 + khalf * 8 + i;
        if (grow < M) C[(size_t)grow * N + gcol] = acc[g][j][i];
      }
    }
  }
}

// ---------------------------------------------------------------------------
// Conversion kernels
// ---------------------------------------------------------------------------
__global__ void f32_to_bf16(const float* __restrict__ in, __bf16* __restrict__ out, int n) {
  int i = (blockIdx.x * blockDim.x + threadIdx.x) * 8;
  if (i < n) {
    v4f a = *(const v4f*)(in + i);
    v4f b = *(const v4f*)(in + i + 4);
    *(v8bf*)(out + i) = cvt8(a, b);
  }
}

// in: [R, C] f32 row-major -> out: [C, R] bf16 row-major (transpose + convert)
__global__ void f32_to_bf16_T(const float* __restrict__ in, __bf16* __restrict__ out,
                              int R, int C) {
  int i = blockIdx.x * blockDim.x + threadIdx.x;
  if (i >= R * C) return;
  int r = i / C, c = i % C;
  out[(size_t)c * R + r] = (__bf16)in[i];
}

// ---------------------------------------------------------------------------
// Attention helpers
// ---------------------------------------------------------------------------
__device__ __forceinline__ unsigned fkey(float f) {
  unsigned b = __float_as_uint(f);
  return (b & 0x80000000u) ? ~b : (b | 0x80000000u);
}
__device__ __forceinline__ float funkey(unsigned k) {
  unsigned b = (k & 0x80000000u) ? (k & 0x7FFFFFFFu) : ~k;
  return __uint_as_float(b);
}

__global__ __launch_bounds__(256)
void attn_dots(const float* __restrict__ h, const float* __restrict__ att_s,
               const float* __restrict__ att_d, float* __restrict__ a_src,
               float* __restrict__ a_dst, int n_nodes, int C) {
  int node = blockIdx.x * (blockDim.x >> 5) + (threadIdx.x >> 5);
  int lane = threadIdx.x & 31;
  if (node >= n_nodes) return;
  const float* row = h + (size_t)node * C;
  float s1 = 0.0f, s2 = 0.0f;
  for (int c = lane; c < C; c += 32) {
    float v = row[c];
    s1 += v * att_s[c];
    s2 += v * att_d[c];
  }
#pragma unroll
  for (int off = 16; off; off >>= 1) {
    s1 += __shfl_xor(s1, off, 32);
    s2 += __shfl_xor(s2, off, 32);
  }
  if (lane == 0) { a_src[node] = s1; a_dst[node] = s2; }
}

__global__ void init_node_bufs(unsigned* __restrict__ mkey,
                               float* __restrict__ denom, int n) {
  int i = blockIdx.x * blockDim.x + threadIdx.x;
  if (i < n) { mkey[i] = 0x007FFFFFu; /* fkey(-inf) */ denom[i] = 0.0f; }
}

__global__ void zero_f32(float* __restrict__ p, int n) {
  int i = blockIdx.x * blockDim.x + threadIdx.x;
  if (i < n) p[i] = 0.0f;
}

__device__ __forceinline__ float leaky(float x) { return x > 0.0f ? x : SLOPE * x; }

__global__ void edge_max(const int* __restrict__ src, const int* __restrict__ dst,
                         const float* __restrict__ a_s, const float* __restrict__ a_d,
                         unsigned* __restrict__ mkey, int E) {
  int e = blockIdx.x * blockDim.x + threadIdx.x;
  if (e >= E) return;
  int d = dst[e];
  float lg = leaky(a_s[src[e]] + a_d[d]);
  atomicMax(&mkey[d], fkey(lg));
}

__global__ void edge_expsum(const int* __restrict__ src, const int* __restrict__ dst,
                            const float* __restrict__ a_s, const float* __restrict__ a_d,
                            const unsigned* __restrict__ mkey,
                            float* __restrict__ denom, float* __restrict__ ex, int E) {
  int e = blockIdx.x * blockDim.x + threadIdx.x;
  if (e >= E) return;
  int d = dst[e];
  float lg = leaky(a_s[src[e]] + a_d[d]);
  float v = __expf(lg - funkey(mkey[d]));
  ex[e] = v;
  atomicAdd(&denom[d], v);
}

// out[dst] += h[src] * (ex[e]/denom[dst])  -- one wave per edge
__global__ __launch_bounds__(256)
void aggregate(const int* __restrict__ src, const int* __restrict__ dst,
               const float* __restrict__ ex, const float* __restrict__ denom,
               const float* __restrict__ h, float* __restrict__ out, int E, int C) {
  int e = blockIdx.x * (blockDim.x >> 5) + (threadIdx.x >> 5);
  int lane = threadIdx.x & 31;
  if (e >= E) return;
  int s = src[e], d = dst[e];
  float alpha = ex[e] / denom[d];
  const float* hrow = h + (size_t)s * C;
  float* orow = out + (size_t)d * C;
  for (int c = lane; c < C; c += 32)
    atomicAdd(&orow[c], hrow[c] * alpha);
}

__global__ void elu_ip(float* __restrict__ p, int n) {
  int i = blockIdx.x * blockDim.x + threadIdx.x;
  if (i < n) { float v = p[i]; p[i] = v > 0.0f ? v : __expf(v) - 1.0f; }
}

__global__ void relu_ip(float* __restrict__ p, int n) {
  int i = blockIdx.x * blockDim.x + threadIdx.x;
  if (i < n) { float v = p[i]; p[i] = v > 0.0f ? v : 0.0f; }
}

// ---------------------------------------------------------------------------
extern "C" void kernel_launch(void* const* d_in, const int* in_sizes, int n_in,
                              void* d_out, int out_size, void* d_ws, size_t ws_size,
                              hipStream_t stream) {
  const float* x     = (const float*)d_in[0];
  const int*   ei    = (const int*)d_in[1];
  const float* W1    = (const float*)d_in[2];
  const float* W2    = (const float*)d_in[3];
  const float* att_s = (const float*)d_in[4];
  const float* att_d = (const float*)d_in[5];
  const int* src = ei;
  const int* dst = ei + N_EDGES;

  float* latent = (float*)d_out;                      // [N_NODES, LAT]
  float* recon  = latent + (size_t)N_NODES * LAT;     // [N_NODES, IN_DIM]

  // ---- Workspace layout ----
  char* w = (char*)d_ws;
  float* h      = (float*)w;                 w += (size_t)N_NODES * HID * 4;   // f32 [N,HID]
  float* aggbuf = (float*)w;                                                   // f32 [N,HID]
  __bf16* xb    = (__bf16*)w;                w += (size_t)N_NODES * HID * 4;   // xb overlays aggbuf
  float* a_src  = (float*)w;                 w += (size_t)N_NODES * 4;
  float* a_dst  = (float*)w;                 w += (size_t)N_NODES * 4;
  unsigned* mkey = (unsigned*)w;             w += (size_t)N_NODES * 4;
  float* denom  = (float*)w;                 w += (size_t)N_NODES * 4;
  float* ex     = (float*)w;                 w += (size_t)N_EDGES * 4;
  __bf16* hb    = (__bf16*)w;                w += (size_t)N_NODES * HID * 2;   // bf16 [N,HID]
  __bf16* latb  = (__bf16*)w;                w += (size_t)N_NODES * LAT * 2;   // bf16 [N,LAT]
  __bf16* W1b   = (__bf16*)w;                w += (size_t)IN_DIM * HID * 2;    // [IN_DIM,HID]
  __bf16* W1Tb  = (__bf16*)w;                w += (size_t)HID * IN_DIM * 2;    // [HID,IN_DIM]
  __bf16* W2b   = (__bf16*)w;                w += (size_t)HID * LAT * 2;       // [HID,LAT]
  __bf16* W2Tb  = (__bf16*)w;                w += (size_t)LAT * HID * 2;       // [LAT,HID]

  const int NHID = N_NODES * HID;
  const int NIN  = N_NODES * IN_DIM;
  const int NLAT = N_NODES * LAT;
  dim3 blk(256);
  const int GR = (N_NODES + 255) / 256;  // GEMM row blocks

  // 0) one-time bf16 conversions (xb overlays aggbuf: only needed for GEMM1)
  f32_to_bf16<<<dim3((NIN / 8 + 255) / 256), blk, 0, stream>>>(x, xb, NIN);
  f32_to_bf16<<<dim3((IN_DIM * HID / 8 + 255) / 256), blk, 0, stream>>>(W1, W1b, IN_DIM * HID);
  f32_to_bf16<<<dim3((HID * LAT / 8 + 255) / 256), blk, 0, stream>>>(W2, W2b, HID * LAT);
  f32_to_bf16_T<<<dim3((IN_DIM * HID + 255) / 256), blk, 0, stream>>>(W1, W1Tb, IN_DIM, HID);
  f32_to_bf16_T<<<dim3((HID * LAT + 255) / 256), blk, 0, stream>>>(W2, W2Tb, HID, LAT);

  // 1) h1 = x @ W1            (B = W1^T stored [HID, IN_DIM])
  gemm_bf16_wmma<<<dim3(GR, HID / 64), blk, 0, stream>>>(xb, W1Tb, h, N_NODES, HID, IN_DIM);

  // 2) attention logits per node
  attn_dots<<<dim3((N_NODES + 7) / 8), blk, 0, stream>>>(h, att_s, att_d, a_src, a_dst, N_NODES, HID);

  // 3) softmax stats (shared by encoder and decoder passes)
  init_node_bufs<<<dim3((N_NODES + 255) / 256), blk, 0, stream>>>(mkey, denom, N_NODES);
  edge_max<<<dim3((N_EDGES + 255) / 256), blk, 0, stream>>>(src, dst, a_src, a_dst, mkey, N_EDGES);
  edge_expsum<<<dim3((N_EDGES + 255) / 256), blk, 0, stream>>>(src, dst, a_src, a_dst, mkey, denom, ex, N_EDGES);

  // 4) h1o = elu(aggregate(h1))   (aggbuf overwrites xb -- xb dead after GEMM1)
  zero_f32<<<dim3((NHID + 255) / 256), blk, 0, stream>>>(aggbuf, NHID);
  aggregate<<<dim3((N_EDGES + 7) / 8), blk, 0, stream>>>(src, dst, ex, denom, h, aggbuf, N_EDGES, HID);
  elu_ip<<<dim3((NHID + 255) / 256), blk, 0, stream>>>(aggbuf, NHID);

  // 5) latent = h1o @ W2      (B = W2^T stored [LAT, HID])
  f32_to_bf16<<<dim3((NHID / 8 + 255) / 256), blk, 0, stream>>>(aggbuf, hb, NHID);
  gemm_bf16_wmma<<<dim3(GR, LAT / 64), blk, 0, stream>>>(hb, W2Tb, latent, N_NODES, LAT, HID);

  // 6) h3 = latent @ W2^T     (B = W2 stored [HID, LAT])
  f32_to_bf16<<<dim3((NLAT / 8 + 255) / 256), blk, 0, stream>>>(latent, latb, NLAT);
  gemm_bf16_wmma<<<dim3(GR, HID / 64), blk, 0, stream>>>(latb, W2b, h, N_NODES, HID, LAT);

  // 7) h3o = relu(aggregate(h3)) using the SAME softmax stats
  zero_f32<<<dim3((NHID + 255) / 256), blk, 0, stream>>>(aggbuf, NHID);
  aggregate<<<dim3((N_EDGES + 7) / 8), blk, 0, stream>>>(src, dst, ex, denom, h, aggbuf, N_EDGES, HID);
  relu_ip<<<dim3((NHID + 255) / 256), blk, 0, stream>>>(aggbuf, NHID);

  // 8) recon = h3o @ W1^T     (B = W1 stored [IN_DIM, HID])
  f32_to_bf16<<<dim3((NHID / 8 + 255) / 256), blk, 0, stream>>>(aggbuf, hb, NHID);
  gemm_bf16_wmma<<<dim3(GR, IN_DIM / 64), blk, 0, stream>>>(hb, W1b, recon, N_NODES, IN_DIM, HID);

  (void)in_sizes; (void)n_in; (void)out_size; (void)ws_size;
}
